// StepWiseLinearTransformerLayer_57724360458872
// MI455X (gfx1250) — matrix operand order repaired and checked by reading, not compiled, not run on hardware
//
#include <hip/hip_runtime.h>
#include <hip/hip_bf16.h>

typedef __attribute__((ext_vector_type(16))) __bf16 v16bf;
typedef __attribute__((ext_vector_type(8)))  __bf16 v8bf;
typedef __attribute__((ext_vector_type(8)))  float  v8f;

#define SLEN   1024
#define BSZ    16
#define DMODEL 1024
#define NHEAD  16
#define DHEAD  64
#define BH     (BSZ * NHEAD)         // 256
#define EDIM   (NHEAD * (3*DHEAD+1)) // 3088
#define MROWS  (SLEN * BSZ)          // 16384
#define PHEAD  196                   // padded per-head stride (16B-aligned quarters)
#define PROW   (NHEAD * PHEAD)       // 3136 floats per (s,b) row

// ---------------------------------------------------------------- fp32 -> bf16
__global__ void cvt_bf16_kernel(const float* __restrict__ in,
                                __bf16* __restrict__ out, int n) {
    int i = blockIdx.x * blockDim.x + threadIdx.x;
    int stride = gridDim.x * blockDim.x;
    for (; i < n; i += stride) out[i] = (__bf16)in[i];
}

// ------------------------------------------------------- WMMA fragment loaders
__device__ inline v16bf load_fragA(const __bf16* __restrict__ A, int ld,
                                   int r0, int k0, int lane) {
    int m = lane & 15, kh = lane >> 4;
    const __bf16* p = A + (size_t)(r0 + m) * ld + k0 + 16 * kh;
    v8bf lo = *(const v8bf*)p;          // K = k0+16kh .. +7
    v8bf hi = *(const v8bf*)(p + 8);    // K = k0+16kh+8 .. +7
    v16bf f;
#pragma unroll
    for (int e = 0; e < 8; ++e) { f[e] = lo[e]; f[e + 8] = hi[e]; }
    return f;
}

__device__ inline v16bf load_fragB(const __bf16* __restrict__ B, int ld,
                                   int n0, int k0, int lane) {
    int n = lane & 15, kh = lane >> 4;
    const __bf16* p = B + (size_t)(n0 + n) * ld + k0 + 16 * kh;
    return *(const v16bf*)p;            // K = k0+16kh .. +15 contiguous
}

// ----------------------------------------------------------------- bf16 GEMM
// C[M][N] (f32) = A[M][K] * B[N][K]^T (+ resid).  256 thr = 8 waves; each wave
// computes (16*MT) x (16*NT); compile-time tiling keeps acc register-resident.
template<int NT, int MT, bool RESID>
__global__ void __launch_bounds__(256)
gemm_bf16_wmma_kernel(const __bf16* __restrict__ A,
                      const __bf16* __restrict__ B,
                      const float* __restrict__ resid,
                      float* __restrict__ C,
                      int N, int K, int col0) {
    int lane = threadIdx.x & 31;
    int wave = threadIdx.x >> 5;
    int r0 = (blockIdx.y * 8 + wave) * (16 * MT);
    int c0 = blockIdx.x * (16 * NT) + col0;

    v8f acc[MT][NT];
#pragma unroll
    for (int m = 0; m < MT; ++m)
#pragma unroll
        for (int t = 0; t < NT; ++t) acc[m][t] = v8f{};

    for (int k0 = 0; k0 < K; k0 += 32) {
        v16bf a[MT];
#pragma unroll
        for (int m = 0; m < MT; ++m)
            a[m] = load_fragA(A, K, r0 + 16 * m, k0, lane);
#pragma unroll
        for (int t = 0; t < NT; ++t) {
            v16bf b = load_fragB(B, K, c0 + 16 * t, k0, lane);
#pragma unroll
            for (int m = 0; m < MT; ++m)
                acc[m][t] = __builtin_amdgcn_wmma_f32_16x16x32_bf16(
                    false, a[m], false, b, (short)0, acc[m][t], false, false);
        }
    }

    int n = lane & 15, mh = lane >> 4;
#pragma unroll
    for (int m = 0; m < MT; ++m)
#pragma unroll
        for (int t = 0; t < NT; ++t) {
            int col = c0 + 16 * t + n;
#pragma unroll
            for (int r = 0; r < 8; ++r) {
                size_t idx = (size_t)(r0 + 16 * m + 8 * mh + r) * N + col;
                float v = acc[m][t][r];
                if (RESID) v += resid[idx];
                C[idx] = v;
            }
        }
}

// --------------------------------------- elementwise prep -> padded layout
// read raw qkvb [MROWS][NHEAD][193]; write qP [MROWS][NHEAD][196]:
// q (elu+1, row-norm) [0,64) ; k likewise [64,128) ; v copy [128,192) ; beta 192
__device__ inline float elup1(float x) { return x > 0.f ? x + 1.f : expf(x); }

__global__ void prep_kernel(const float* __restrict__ qkvb,
                            float* __restrict__ qP) {
    int R    = blockIdx.x * 8 + (threadIdx.x >> 5);   // 0 .. MROWS*NHEAD-1
    int lane = threadIdx.x & 31;
    int m = R >> 4, hd = R & 15;
    size_t base  = (size_t)m * EDIM + (size_t)hd * 193;
    size_t base2 = (size_t)m * PROW + (size_t)hd * PHEAD;

    float a0 = elup1(qkvb[base + lane]);
    float a1 = elup1(qkvb[base + 32 + lane]);
    float s = a0 + a1;
#pragma unroll
    for (int off = 16; off; off >>= 1) s += __shfl_xor(s, off, 32);
    qP[base2 + lane]      = a0 / s;
    qP[base2 + 32 + lane] = a1 / s;

    float b0 = elup1(qkvb[base + 64 + lane]);
    float b1 = elup1(qkvb[base + 96 + lane]);
    float sk = b0 + b1;
#pragma unroll
    for (int off = 16; off; off >>= 1) sk += __shfl_xor(sk, off, 32);
    qP[base2 + 64 + lane] = b0 / sk;
    qP[base2 + 96 + lane] = b1 / sk;

    qP[base2 + 128 + lane]      = qkvb[base + 128 + lane];
    qP[base2 + 160 + lane]      = qkvb[base + 160 + lane];

    if (lane == 0) {
        float x = qkvb[base + 192];
        qP[base2 + 192] = 1.f / (1.f + expf(-x));
    }
}

// -------------------------------------------------------- delta-rule scan
// One block (256 thr = 8 waves) per head.  Thread (row, quarter) owns
// W[row][16*quarter .. +15] in VGPRs.  Quarters of a row are adjacent lanes
// -> 4-lane shfl_xor reductions, no LDS, no barriers.  k/q/v/beta for step
// s+1 are register double-buffered; step s+3 is prefetched.
__device__ __forceinline__ float rowsum4(float x) {
    x += __shfl_xor(x, 1, 32);
    x += __shfl_xor(x, 2, 32);
    return x;
}

__global__ void __launch_bounds__(256)
scan_kernel(const float* __restrict__ qP, __bf16* __restrict__ lo) {
    int bh = blockIdx.x;             // 0..255
    int b  = bh >> 4, hd = bh & 15;
    int tid  = threadIdx.x;
    int row  = tid >> 2;             // 0..63
    int quar = tid & 3;              // 0..3

    const float* hb = qP + (size_t)hd * PHEAD;

    float W[16];
#pragma unroll
    for (int c = 0; c < 16; ++c) W[c] = 0.f;

    float4 kA[4], qA[4], kB[4], qB[4];
    float vA, bA, vB, bB;

    auto LD = [&](int s, float4* kk, float4* qq, float& vv, float& be) {
        const float* p = hb + (size_t)(s * BSZ + b) * PROW;
        const float4* kp = (const float4*)(p + 64 + 16 * quar);
        const float4* qp = (const float4*)(p + 16 * quar);
#pragma unroll
        for (int j = 0; j < 4; ++j) { kk[j] = kp[j]; qq[j] = qp[j]; }
        vv = p[128 + row];
        be = p[192];
    };

    auto STEP = [&](int s, const float4* kk, const float4* qq,
                    float vv, float be) {
        float vo0 = 0.f, vo1 = 0.f;
#pragma unroll
        for (int j = 0; j < 4; ++j) {
            float4 K = kk[j];
            vo0 += W[4*j+0] * K.x + W[4*j+2] * K.z;
            vo1 += W[4*j+1] * K.y + W[4*j+3] * K.w;
        }
        float vo = rowsum4(vo0 + vo1);
        float vins = be * (vv - vo);

        float o0 = 0.f, o1 = 0.f;
#pragma unroll
        for (int j = 0; j < 4; ++j) {
            float4 K = kk[j];
            float4 Q = qq[j];
            W[4*j+0] += vins * K.x;  o0 += W[4*j+0] * Q.x;
            W[4*j+1] += vins * K.y;  o1 += W[4*j+1] * Q.y;
            W[4*j+2] += vins * K.z;  o0 += W[4*j+2] * Q.z;
            W[4*j+3] += vins * K.w;  o1 += W[4*j+3] * Q.w;
        }
        float o = rowsum4(o0 + o1);
        if (quar == 0)
            lo[((size_t)(s * BSZ + b)) * DMODEL + hd * DHEAD + row] =
                (__bf16)(o * 0.125f);   // 1/sqrt(64)
    };

    LD(0, kA, qA, vA, bA);
    for (int s = 0; s < SLEN; s += 2) {
        LD(s + 1, kB, qB, vB, bB);
        int sp = (s + 3 < SLEN) ? s + 3 : SLEN - 1;
        __builtin_prefetch(hb + (size_t)(sp * BSZ + b) * PROW, 0, 3);
        STEP(s, kA, qA, vA, bA);
        if (s + 2 < SLEN) LD(s + 2, kA, qA, vA, bA);   // uniform branch
        STEP(s + 1, kB, qB, vB, bB);
    }
}

// ---------------------------------------------------------------- LayerNorm
__global__ void ln_kernel(const float* __restrict__ res,
                          const float* __restrict__ g,
                          const float* __restrict__ bb,
                          float* __restrict__ out) {
    int m = blockIdx.x;
    size_t base = (size_t)m * DMODEL;
    int t = threadIdx.x, lane = t & 31, wid = t >> 5;
    __shared__ float sm[8];
    __shared__ float bc;

    float x[4]; float s = 0.f;
#pragma unroll
    for (int j = 0; j < 4; ++j) { x[j] = res[base + t + 256 * j]; s += x[j]; }
#pragma unroll
    for (int off = 16; off; off >>= 1) s += __shfl_xor(s, off, 32);
    if (!lane) sm[wid] = s;
    __syncthreads();
    if (t == 0) { float tot = 0.f; for (int q = 0; q < 8; ++q) tot += sm[q];
                  bc = tot * (1.0f / DMODEL); }
    __syncthreads();
    float mu = bc;
    float v = 0.f;
#pragma unroll
    for (int j = 0; j < 4; ++j) { float d = x[j] - mu; v += d * d; }
#pragma unroll
    for (int off = 16; off; off >>= 1) v += __shfl_xor(v, off, 32);
    __syncthreads();
    if (!lane) sm[wid] = v;
    __syncthreads();
    if (t == 0) { float tot = 0.f; for (int q = 0; q < 8; ++q) tot += sm[q];
                  bc = rsqrtf(tot * (1.0f / DMODEL) + 1e-5f); }
    __syncthreads();
    float rs = bc;
#pragma unroll
    for (int j = 0; j < 4; ++j) {
        int col = t + 256 * j;
        out[base + col] = (x[j] - mu) * rs * g[col] + bb[col];
    }
}

// ------------------------------------------------------------------- launch
extern "C" void kernel_launch(void* const* d_in, const int* in_sizes, int n_in,
                              void* d_out, int out_size, void* d_ws, size_t ws_size,
                              hipStream_t stream) {
    const float* h   = (const float*)d_in[0];   // [1024][16][1024]
    const float* Wq  = (const float*)d_in[1];   // [3088][1024]
    const float* Wo  = (const float*)d_in[2];   // [1024][1024]
    const float* lng = (const float*)d_in[3];
    const float* lnb = (const float*)d_in[4];
    float* out = (float*)d_out;

    const size_t N_H  = (size_t)MROWS * DMODEL;   // 16,777,216
    const size_t N_WQ = (size_t)EDIM  * DMODEL;   //  3,162,112
    const size_t N_WO = (size_t)DMODEL * DMODEL;  //  1,048,576

    char* ws = (char*)d_ws;
    const size_t OFF_HB   = 0;                        // bf16 h        32 MB
    const size_t OFF_WQB  = OFF_HB  + N_H  * 2;       // bf16 W_qkvb  ~6 MB
    const size_t OFF_WOB  = OFF_WQB + N_WQ * 2;       // bf16 W_o      2 MB
    const size_t OFF_QKVB = OFF_WOB + N_WO * 2;       // raw qkvb    193 MB
    // raw qkvb is dead after prep_kernel: alias RES and LO into it
    const size_t OFF_RES  = OFF_QKVB;                 // f32 h+attn   64 MB
    const size_t OFF_LO   = OFF_RES + N_H * 4;        // bf16 lo      32 MB
    const size_t OFF_QP   = OFF_QKVB + (size_t)MROWS * EDIM * 4;  // 205 MB

    __bf16* hB   = (__bf16*)(ws + OFF_HB);
    __bf16* wqB  = (__bf16*)(ws + OFF_WQB);
    __bf16* woB  = (__bf16*)(ws + OFF_WOB);
    float*  qkvb = (float*) (ws + OFF_QKVB);
    float*  qP   = (float*) (ws + OFF_QP);
    __bf16* loB  = (__bf16*)(ws + OFF_LO);
    float*  res  = (float*) (ws + OFF_RES);

    // 1) fp32 -> bf16 for WMMA operands
    cvt_bf16_kernel<<<4096, 256, 0, stream>>>(h,  hB,  (int)N_H);
    cvt_bf16_kernel<<<2048, 256, 0, stream>>>(Wq, wqB, (int)N_WQ);
    cvt_bf16_kernel<<<1024, 256, 0, stream>>>(Wo, woB, (int)N_WO);

    // 2) qkvb = h @ W_qkvb^T  (M=16384, N=3088, K=1024)
    {
        dim3 gMain(48, MROWS / 256);
        gemm_bf16_wmma_kernel<4, 2, false><<<gMain, 256, 0, stream>>>(
            hB, wqB, nullptr, qkvb, EDIM, DMODEL, 0);
        dim3 gTail(1, MROWS / 256);
        gemm_bf16_wmma_kernel<1, 2, false><<<gTail, 256, 0, stream>>>(
            hB, wqB, nullptr, qkvb, EDIM, DMODEL, 3072);
    }

    // 3) elu+1 / row-normalize q,k ; sigmoid beta ; repack padded
    prep_kernel<<<(MROWS * NHEAD) / 8, 256, 0, stream>>>(qkvb, qP);

    // 4) sequential delta-rule scan, one block (256 thr) per head
    scan_kernel<<<BH, 256, 0, stream>>>(qP, loB);

    // 5) res = layer_out @ W_o^T + h   (M=16384, N=1024, K=1024)
    {
        dim3 gD(DMODEL / 64, MROWS / 256);
        gemm_bf16_wmma_kernel<4, 2, true><<<gD, 256, 0, stream>>>(
            loB, woB, h, res, DMODEL, DMODEL, 0);
    }

    // 6) LayerNorm
    ln_kernel<<<MROWS, 256, 0, stream>>>(res, lng, lnb, out);
}
